// HMM_369367188126
// MI455X (gfx1250) — compile-verified, matrix operand order checked by reading, not run on hardware
//
#include <hip/hip_runtime.h>
#include <math.h>

// ---------------------------------------------------------------------------
// Log-space HMM + DFA decode for MI455X (gfx1250, wave32, WMMA).
// All heavy matmuls run in the linear (exp) domain in f16 via
// v_wmma_f32_16x16x32_f16 with f32 accumulation; log/bias epilogues restore
// the log domain.
//
// Operand packing: the 16x16x32 f16 fragment layout groups K indices as
// {0-7,16-23} (lanes 0-15) and {8-15,24-31} (lanes 16-31) -- i.e. a swap of
// bits 3 and 4 of k (involution pk). Every GEMM operand is pre-stored with
// this permutation (A row-major [M,K]pk, B column-major [N,K]pk), so LDS
// staging is contiguous b128 copies and each fragment is ONE aligned 32-byte
// LDS load. Workspace budget ~490 MB carved from d_ws.
// ---------------------------------------------------------------------------

#define Hd   2048
#define Vd   32000
#define Sd   64
#define Ed   512
#define Bd   8
#define Pd   16
#define Ld   8
#define Tp1d 33
#define BSd  (Bd * Sd)

#define NEGV (-1e30f)
#define POSBIG (3.4e38f)

typedef _Float16 h16;
typedef __attribute__((ext_vector_type(16))) _Float16 v16h;
typedef __attribute__((ext_vector_type(8)))  float    v8f;

#define EPI_PLAIN   0
#define EPI_LOG     1   // out = log(acc) + rowBias + colBias (clamped)
#define EPI_LOG_N2N 2   // same + nan_to_num semantics

// fragment k-permutation: swap bit3<->bit4 within each 32-element K group
__device__ __forceinline__ int pkk(int k) {
    return (k & ~31) | (k & 7) | ((k & 8) << 1) | ((k & 16) >> 1);
}

// ---------------------------------------------------------------------------
// WMMA f16 GEMM: C[M,N] = epilogue(A @ B)
// Apk: row-major [M,K] with pk() applied within each 32-k group.
// Bpk: column-major [N,K] with pk() applied within each 32-k group.
// Block = 128 threads = 4 waves; tile 64(M) x 64(N), BK = 64 (8 WMMA/stage).
// Requires M%64==0, N%64==0, K%64==0 (true for all call sites).
// ---------------------------------------------------------------------------
__global__ __launch_bounds__(128)
void k_gemm_wmma(const h16* __restrict__ Apk, const h16* __restrict__ Bpk,
                 float* __restrict__ C, int M, int N, int K,
                 const float* __restrict__ rowBias,
                 const float* __restrict__ colBias, int mode)
{
    __shared__ alignas(32) h16 As[64 * 80];   // pitch 80 halves = 160B
    __shared__ alignas(32) h16 Bs[64 * 80];

    const int tid  = threadIdx.x;
    const int lane = tid & 31;
    const int wave = tid >> 5;          // 0..3
    const int wm   = wave >> 1;         // 32-row half
    const int wn   = wave & 1;          // 32-col half
    const int m0   = blockIdx.y * 64;
    const int n0   = blockIdx.x * 64;

    v8f acc[2][2] = {};

    const int srow  = tid >> 1;         // 0..63 (A row index / B col index)
    const int shalf = (tid & 1) * 32;   // k-offset 0 or 32 within BK

    const int r   = lane & 15;
    const int kb2 = (lane & 16) ? 16 : 0;  // fragment half within packed 32

    for (int k0 = 0; k0 < K; k0 += 64) {
        // contiguous 64B copies (global b128 -> LDS b128), layouts pre-packed
        {
            const uint4* ga = (const uint4*)(Apk + (size_t)(m0 + srow) * K + k0 + shalf);
            uint4* la = (uint4*)(As + srow * 80 + shalf);
#pragma unroll
            for (int j = 0; j < 4; ++j) la[j] = ga[j];
        }
        {
            const uint4* gb = (const uint4*)(Bpk + (size_t)(n0 + srow) * K + k0 + shalf);
            uint4* lb = (uint4*)(Bs + srow * 80 + shalf);
#pragma unroll
            for (int j = 0; j < 4; ++j) lb[j] = gb[j];
        }
        if (k0 + 64 < K) {  // speculative prefetch of next tiles
            __builtin_prefetch(Apk + (size_t)(m0 + srow) * K + k0 + 64 + shalf, 0, 1);
            __builtin_prefetch(Bpk + (size_t)(n0 + srow) * K + k0 + 64 + shalf, 0, 1);
        }
        __syncthreads();

#pragma unroll
        for (int c = 0; c < 2; ++c) {
            v16h af[2], bf[2];
#pragma unroll
            for (int mi = 0; mi < 2; ++mi)
                af[mi] = *(const v16h*)(As + (wm * 32 + mi * 16 + r) * 80 + c * 32 + kb2);
#pragma unroll
            for (int ni = 0; ni < 2; ++ni)
                bf[ni] = *(const v16h*)(Bs + (wn * 32 + ni * 16 + r) * 80 + c * 32 + kb2);
#pragma unroll
            for (int mi = 0; mi < 2; ++mi)
#pragma unroll
                for (int ni = 0; ni < 2; ++ni)
                    acc[mi][ni] = __builtin_amdgcn_wmma_f32_16x16x32_f16(
                        false, af[mi], false, bf[ni], (short)0, acc[mi][ni],
                        false, false);
        }
        __syncthreads();
    }

    // C/D layout: VGPR rr -> (M = rr + (lane>=16 ? 8 : 0), N = lane & 15)
    const int cn   = lane & 15;
    const int moff = (lane & 16) ? 8 : 0;
#pragma unroll
    for (int mi = 0; mi < 2; ++mi) {
#pragma unroll
        for (int ni = 0; ni < 2; ++ni) {
#pragma unroll
            for (int rr = 0; rr < 8; ++rr) {
                const int gm = m0 + wm * 32 + mi * 16 + moff + rr;
                const int gn = n0 + wn * 32 + ni * 16 + cn;
                float v = acc[mi][ni][rr];
                if (mode != EPI_PLAIN) {
                    float r2 = (v > 0.f) ? logf(v) : NEGV;
                    if (rowBias) r2 += rowBias[gm];
                    if (colBias) r2 += colBias[gn];
                    if (mode == EPI_LOG_N2N) {
                        if (isnan(r2))      r2 = 0.f;
                        else if (isinf(r2)) r2 = (r2 < 0.f) ? NEGV : POSBIG;
                    } else {
                        if (r2 < NEGV) r2 = NEGV;
                    }
                    v = r2;
                }
                C[(size_t)gm * N + gn] = v;
            }
        }
    }
}

// ----------------------------- setup kernels -------------------------------

// per-row max + log-sum-exp of beta_raw (rowStat = max + log(sum exp(x-max)))
__global__ void k_row_stats(const float* __restrict__ braw,
                            float* __restrict__ rowmax,
                            float* __restrict__ rowStat)
{
    __shared__ float red[256];
    const int h = blockIdx.x, t = threadIdx.x;
    const float* row = braw + (size_t)h * Vd;
    float mx = -INFINITY;
    for (int v = t; v < Vd; v += 256) mx = fmaxf(mx, row[v]);
    red[t] = mx; __syncthreads();
    for (int s = 128; s > 0; s >>= 1) { if (t < s) red[t] = fmaxf(red[t], red[t + s]); __syncthreads(); }
    mx = red[0]; __syncthreads();
    float sm = 0.f;
    for (int v = t; v < Vd; v += 256) sm += expf(row[v] - mx);
    red[t] = sm; __syncthreads();
    for (int s = 128; s > 0; s >>= 1) { if (t < s) red[t] += red[t + s]; __syncthreads(); }
    if (t == 0) { rowmax[h] = mx; rowStat[h] = mx + logf(red[0]); }
}

__global__ void k_gamma(const float* __restrict__ g, float* __restrict__ out)
{
    __shared__ float red[256];
    const int t = threadIdx.x;
    float mx = -INFINITY;
    for (int h = t; h < Hd; h += 256) mx = fmaxf(mx, g[h]);
    red[t] = mx; __syncthreads();
    for (int s = 128; s > 0; s >>= 1) { if (t < s) red[t] = fmaxf(red[t], red[t + s]); __syncthreads(); }
    mx = red[0]; __syncthreads();
    float sm = 0.f;
    for (int h = t; h < Hd; h += 256) sm += expf(g[h] - mx);
    red[t] = sm; __syncthreads();
    for (int s = 128; s > 0; s >>= 1) { if (t < s) red[t] += red[t + s]; __syncthreads(); }
    const float z = mx + logf(red[0]); __syncthreads();
    for (int h = t; h < Hd; h += 256) out[h] = g[h] - z;
}

// row-softmax of alpha -> alphaExp (f32) and packed B-operand of alpha^T:
// aTpk[n*H + pk(k)] = alphaExp[n][k]  (block row h == n, j == k)
__global__ void k_alpha(const float* __restrict__ a, float* __restrict__ aexp,
                        h16* __restrict__ aTpk)
{
    __shared__ float red[256];
    const int h = blockIdx.x, t = threadIdx.x;
    const float* row = a + (size_t)h * Hd;
    float mx = -INFINITY;
    for (int j = t; j < Hd; j += 256) mx = fmaxf(mx, row[j]);
    red[t] = mx; __syncthreads();
    for (int s = 128; s > 0; s >>= 1) { if (t < s) red[t] = fmaxf(red[t], red[t + s]); __syncthreads(); }
    mx = red[0]; __syncthreads();
    float sm = 0.f;
    for (int j = t; j < Hd; j += 256) sm += expf(row[j] - mx);
    red[t] = sm; __syncthreads();
    for (int s = 128; s > 0; s >>= 1) { if (t < s) red[t] += red[t + s]; __syncthreads(); }
    const float inv = 1.f / red[0];
    for (int j = t; j < Hd; j += 256) {
        const float e = expf(row[j] - mx) * inv;
        aexp[(size_t)h * Hd + j] = e;
        aTpk[(size_t)h * Hd + pkk(j)] = (h16)e;
    }
}

// colmax over h of beta = braw - rowStat
__global__ void k_colmaxB(const float* __restrict__ braw,
                          const float* __restrict__ rowStat,
                          float* __restrict__ cmax)
{
    __shared__ float rs[Hd];
    for (int h = threadIdx.x; h < Hd; h += 256) rs[h] = rowStat[h];
    __syncthreads();
    const int v = blockIdx.x * 256 + threadIdx.x;
    if (v >= Vd) return;
    float mx = -INFINITY;
    for (int h = 0; h < Hd; ++h) mx = fmaxf(mx, braw[(size_t)h * Vd + v] - rs[h]);
    cmax[v] = mx;
}

// Ppk (B-operand of T_weights GEMM, [N=H][K=V] packed):
// Ppk[h*V + pk(v)] = exp(braw[h,v] - rowmax[h])
__global__ void k_packPtw(const float* __restrict__ braw,
                          const float* __restrict__ rowmax, h16* __restrict__ P)
{
    const size_t idx = (size_t)blockIdx.x * 256 + threadIdx.x;
    if (idx >= (size_t)Hd * Vd) return;
    const int h = (int)(idx / Vd), v = (int)(idx % Vd);
    P[(size_t)h * Vd + pkk(v)] = (h16)expf(braw[idx] - rowmax[h]);
}

// expBpk (B-operand of logits GEMM, [N=V][K=H] packed), tiled transpose:
// expBpk[v*H + pk(h)] = exp(braw[h,v] - rowStat[h] - cmax[v])
__global__ void k_packExpB(const float* __restrict__ braw,
                           const float* __restrict__ rowStat,
                           const float* __restrict__ cmax, h16* __restrict__ out)
{
    __shared__ h16 tile[32][33];
    const int h0 = blockIdx.x * 32, v0 = blockIdx.y * 32;
    const int tx = threadIdx.x, ty = threadIdx.y; // 32x8
#pragma unroll
    for (int i = 0; i < 4; ++i) {
        const int hh = h0 + ty + i * 8;
        tile[ty + i * 8][tx] =
            (h16)expf(braw[(size_t)hh * Vd + v0 + tx] - rowStat[hh] - cmax[v0 + tx]);
    }
    __syncthreads();
#pragma unroll
    for (int i = 0; i < 4; ++i) {
        const int vv = v0 + ty + i * 8;
        out[(size_t)vv * Hd + h0 + pkk(tx)] = tile[tx][ty + i * 8];
    }
}

// Tmask as B-operand ([N=V][K=E] packed), tiled transpose of T_mask[E,V]
__global__ void k_packMaskT(const float* __restrict__ src, h16* __restrict__ dst)
{
    __shared__ h16 tile[32][33];
    const int e0 = blockIdx.x * 32, v0 = blockIdx.y * 32;
    const int tx = threadIdx.x, ty = threadIdx.y; // 32x8
#pragma unroll
    for (int i = 0; i < 4; ++i)
        tile[ty + i * 8][tx] = (h16)src[(size_t)(e0 + ty + i * 8) * Vd + v0 + tx];
    __syncthreads();
#pragma unroll
    for (int i = 0; i < 4; ++i)
        dst[(size_t)(v0 + ty + i * 8) * Ed + e0 + pkk(tx)] = tile[tx][ty + i * 8];
}

// generic A-operand packer: dst[row*cols + pk(c)] = (h16)src[row*cols + c]
__global__ void k_packA(const float* __restrict__ src, h16* __restrict__ dst,
                        int rows, int cols)
{
    const int idx = blockIdx.x * 256 + threadIdx.x;
    if (idx >= rows * cols) return;
    const int rI = idx / cols, c = idx % cols;
    dst[(size_t)rI * cols + pkk(c)] = (h16)src[idx];
}

__global__ void k_bmT(const float* __restrict__ rowmax,
                      const float* __restrict__ rowStat, float* __restrict__ bm)
{
    const int h = blockIdx.x * 256 + threadIdx.x;
    if (h < Hd) bm[h] = rowmax[h] - rowStat[h];
}

// ----------------------------- sequential scans ----------------------------

__global__ __launch_bounds__(1024)
void k_prefix(const float* __restrict__ aexp, const float* __restrict__ braw,
              const float* __restrict__ rowStat, const float* __restrict__ gam,
              const int* __restrict__ pref, float* __restrict__ Acache)
{
    __shared__ float y[Hd];
    __shared__ float e[Hd];
    __shared__ float red[1024];
    const int b = blockIdx.x, t = threadIdx.x;
    for (int h = t; h < Hd; h += 1024) y[h] = gam[h];
    __syncthreads();
    for (int p = 0; p < Pd; ++p) {
        const int tok = pref[b * Pd + p];
        for (int h = t; h < Hd; h += 1024)
            y[h] += braw[(size_t)h * Vd + tok] - rowStat[h];
        __syncthreads();
        float mx = -INFINITY;
        for (int h = t; h < Hd; h += 1024) mx = fmaxf(mx, y[h]);
        red[t] = mx; __syncthreads();
        for (int s = 512; s > 0; s >>= 1) { if (t < s) red[t] = fmaxf(red[t], red[t + s]); __syncthreads(); }
        const float m = red[0]; __syncthreads();
        for (int h = t; h < Hd; h += 1024) e[h] = expf(y[h] - m);
        __syncthreads();
        for (int h = t; h < Hd; h += 1024) {
            float acc = 0.f;
            for (int k = 0; k < Hd; ++k) acc += e[k] * aexp[(size_t)k * Hd + h];
            y[h] = logf(acc) + m;
        }
        __syncthreads();
    }
    for (int h = t; h < Hd; h += 1024) Acache[(size_t)b * Hd + h] = y[h];
}

__global__ __launch_bounds__(1024)
void k_suffix(const h16* __restrict__ aTpk, const float* __restrict__ braw,
              const float* __restrict__ rowStat, const int* __restrict__ suf,
              float* __restrict__ ysuf, float* __restrict__ ysufmax)
{
    __shared__ float y[Hd];
    __shared__ float e[Hd];
    __shared__ float red[1024];
    const int t = threadIdx.x;
    const int tokL = suf[Ld - 1];
    for (int h = t; h < Hd; h += 1024)
        y[h] = braw[(size_t)h * Vd + tokL] - rowStat[h];
    __syncthreads();
    for (int i = Ld - 2; i >= 0; --i) {
        const int tok = suf[i];
        float mx = -INFINITY;
        for (int h = t; h < Hd; h += 1024) mx = fmaxf(mx, y[h]);
        red[t] = mx; __syncthreads();
        for (int s = 512; s > 0; s >>= 1) { if (t < s) red[t] = fmaxf(red[t], red[t + s]); __syncthreads(); }
        const float m = red[0]; __syncthreads();
        for (int h = t; h < Hd; h += 1024) e[h] = expf(y[h] - m);
        __syncthreads();
        for (int h = t; h < Hd; h += 1024) {
            float acc = 0.f;
            // sum_k alpha[h][k]*e[k]; aTpk[h*H + i] = alpha[h][pk(i)], pk involutive
            for (int k = 0; k < Hd; ++k)
                acc += (float)aTpk[(size_t)h * Hd + k] * e[pkk(k)];
            y[h] = logf(acc) + m + braw[(size_t)h * Vd + tok] - rowStat[h];
        }
        __syncthreads();
    }
    float mx = -INFINITY;
    for (int h = t; h < Hd; h += 1024) { ysuf[h] = y[h]; mx = fmaxf(mx, y[h]); }
    red[t] = mx; __syncthreads();
    for (int s = 512; s > 0; s >>= 1) { if (t < s) red[t] = fmaxf(red[t], red[t + s]); __syncthreads(); }
    if (t == 0) ysufmax[0] = red[0];
}

// ----------------------------- c_step helpers ------------------------------

// y0 prep: E3pk[s*H + pk(h)] = exp(y_[s,h] - Am[s]) (A-operand packed)
__global__ void k_y0prep(const float* __restrict__ ysuf,
                         const float* __restrict__ ysufmax,
                         const float* __restrict__ accept,
                         float* __restrict__ Am, h16* __restrict__ E3)
{
    const int s = blockIdx.x;
    const bool on = accept[s] > 0.f;
    const float am = on ? ysufmax[0] : NEGV;
    if (threadIdx.x == 0) Am[s] = am;
    for (int h = threadIdx.x; h < Hd; h += 256) {
        const float yv = on ? ysuf[h] : NEGV;
        E3[(size_t)s * Hd + pkk(h)] = (h16)expf(yv - am);
    }
}

// colmax over R rows of X(+add) -> Bm[H]; Epk[h*R + pk(r)] = exp(x - Bm[h])
// (B-operand packed, column-major)
__global__ void k_colmax_exp(const float* __restrict__ X,
                             const float* __restrict__ add,
                             float* __restrict__ Bm, h16* __restrict__ E,
                             int R)
{
    const int h = blockIdx.x * 256 + threadIdx.x;
    if (h >= Hd) return;
    float mx = -INFINITY;
    for (int rI = 0; rI < R; ++rI) {
        float v = X[(size_t)rI * Hd + h];
        if (add) v += add[(size_t)rI * Hd + h];
        mx = fmaxf(mx, v);
    }
    Bm[h] = mx;
    for (int rI = 0; rI < R; ++rI) {
        float v = X[(size_t)rI * Hd + h];
        if (add) v += add[(size_t)rI * Hd + h];
        E[(size_t)h * R + pkk(rI)] = (h16)expf(v - mx);
    }
}

// rowmax over H per row -> Am[R]; Epk[r*H + pk(h)] = exp(x - Am[r]) (A-operand)
__global__ void k_rowmax_exp(const float* __restrict__ X, float* __restrict__ Am,
                             h16* __restrict__ E)
{
    __shared__ float red[256];
    const int rI = blockIdx.x, t = threadIdx.x;
    float mx = -INFINITY;
    for (int h = t; h < Hd; h += 256) mx = fmaxf(mx, X[(size_t)rI * Hd + h]);
    red[t] = mx; __syncthreads();
    for (int s = 128; s > 0; s >>= 1) { if (t < s) red[t] = fmaxf(red[t], red[t + s]); __syncthreads(); }
    mx = red[0];
    if (t == 0) Am[rI] = mx;
    __syncthreads();
    for (int h = t; h < Hd; h += 256)
        E[(size_t)rI * Hd + pkk(h)] = (h16)expf(X[(size_t)rI * Hd + h] - mx);
}

// range aggregation: Cr[b,s,h] = n2n(log(sum_t rm[b,t]*exp(C[t,s,h]-max))) + max
__global__ void k_range(const float* __restrict__ Cbuf,
                        const float* __restrict__ rm, float* __restrict__ Cr)
{
    const int idx = blockIdx.x * 256 + threadIdx.x;  // over S*H
    if (idx >= Sd * Hd) return;
    float mx = -INFINITY;
    for (int t = 0; t < Tp1d; ++t)
        mx = fmaxf(mx, Cbuf[(size_t)t * Sd * Hd + idx]);
    for (int b = 0; b < Bd; ++b) {
        float sm = 0.f;
        for (int t = 0; t < Tp1d; ++t)
            if (rm[b * Tp1d + t] > 0.f)
                sm += expf(Cbuf[(size_t)t * Sd * Hd + idx] - mx);
        float r = (sm > 0.f ? logf(sm) : NEGV) + mx;
        if (isnan(r)) r = 0.f;
        else if (isinf(r)) r = (r < 0.f) ? NEGV : POSBIG;
        else if (r < NEGV) r = NEGV;
        Cr[(size_t)b * Sd * Hd + idx] = r;
    }
}

// Cfull = A[b] + Cr[b,s]; rowmax + exp f16 (A-operand packed) per (b,s) row
__global__ void k_cfull_prep(const float* __restrict__ Acache,
                             const float* __restrict__ Cr,
                             float* __restrict__ AmC, h16* __restrict__ expC)
{
    __shared__ float red[256];
    const int bs = blockIdx.x, t = threadIdx.x;
    const int b = bs / Sd;
    float mx = -INFINITY;
    for (int h = t; h < Hd; h += 256)
        mx = fmaxf(mx, Acache[(size_t)b * Hd + h] + Cr[(size_t)bs * Hd + h]);
    red[t] = mx; __syncthreads();
    for (int s = 128; s > 0; s >>= 1) { if (t < s) red[t] = fmaxf(red[t], red[t + s]); __syncthreads(); }
    mx = red[0];
    if (t == 0) AmC[bs] = mx;
    __syncthreads();
    for (int h = t; h < Hd; h += 256)
        expC[(size_t)bs * Hd + pkk(h)] =
            (h16)expf(Acache[(size_t)b * Hd + h] + Cr[(size_t)bs * Hd + h] - mx);
}

// M3pk[bs*E + pk(e)] = VE[states[b],e] * EV[e,s]  (A-operand packed)
__global__ void k_m3(const float* __restrict__ VE, const float* __restrict__ EV,
                     const int* __restrict__ states, h16* __restrict__ M3)
{
    const int idx = blockIdx.x * 256 + threadIdx.x;
    if (idx >= BSd * Ed) return;
    const int bs = idx / Ed, e = idx % Ed;
    const int b = bs / Sd, s = bs % Sd;
    const int st = states[b];
    M3[(size_t)bs * Ed + pkk(e)] =
        (h16)(VE[(size_t)st * Ed + e] * EV[(size_t)e * Sd + s]);
}

// logits[b,v] = logsumexp_s( lp[b,s,v] + log(m[b,s,v]) )  (m==0 -> NEG)
__global__ void k_final(const float* __restrict__ lp, const float* __restrict__ mW,
                        float* __restrict__ out)
{
    const int idx = blockIdx.x * 256 + threadIdx.x;
    if (idx >= Bd * Vd) return;
    const int b = idx / Vd, v = idx % Vd;
    float mx = -INFINITY;
    for (int s = 0; s < Sd; ++s) {
        const size_t o = (size_t)(b * Sd + s) * Vd + v;
        const float mv = mW[o];
        const float val = lp[o] + (mv > 0.f ? logf(mv) : NEGV);
        mx = fmaxf(mx, val);
    }
    float sm = 0.f;
    for (int s = 0; s < Sd; ++s) {
        const size_t o = (size_t)(b * Sd + s) * Vd + v;
        const float mv = mW[o];
        const float val = lp[o] + (mv > 0.f ? logf(mv) : NEGV);
        sm += expf(val - mx);
    }
    out[idx] = mx + logf(sm);
}

__global__ void k_prepA(const float* __restrict__ Acache, float* __restrict__ rmax,
                        float* __restrict__ expA)
{
    __shared__ float red[256];
    const int b = blockIdx.x, t = threadIdx.x;
    float mx = -INFINITY;
    for (int h = t; h < Hd; h += 256) mx = fmaxf(mx, Acache[(size_t)b * Hd + h]);
    red[t] = mx; __syncthreads();
    for (int s = 128; s > 0; s >>= 1) { if (t < s) red[t] = fmaxf(red[t], red[t + s]); __syncthreads(); }
    mx = red[0];
    if (t == 0) rmax[b] = mx;
    __syncthreads();
    for (int h = t; h < Hd; h += 256)
        expA[(size_t)b * Hd + h] = expf(Acache[(size_t)b * Hd + h] - mx);
}

// logits_[b,v] = log( sum_h expA[b,h]*exp(beta[h,v]-cmax[v]) ) + rmaxA[b]+cmax[v]
__global__ void k_logits2(const float* __restrict__ expA,
                          const float* __restrict__ rmaxA,
                          const float* __restrict__ braw,
                          const float* __restrict__ rowStat,
                          const float* __restrict__ cmax, float* __restrict__ out)
{
    __shared__ float rs[Hd];
    for (int h = threadIdx.x; h < Hd; h += 256) rs[h] = rowStat[h];
    __syncthreads();
    const int v = blockIdx.x * 256 + threadIdx.x;
    if (v >= Vd) return;
    const float cm = cmax[v];
    float acc[Bd] = {};
    for (int h = 0; h < Hd; ++h) {
        const float eb = expf(braw[(size_t)h * Vd + v] - rs[h] - cm);
#pragma unroll
        for (int b = 0; b < Bd; ++b) acc[b] += expA[(size_t)b * Hd + h] * eb;
    }
#pragma unroll
    for (int b = 0; b < Bd; ++b) {
        const float r = (acc[b] > 0.f ? logf(acc[b]) : NEGV) + rmaxA[b] + cm;
        out[(size_t)b * Vd + v] = r;
    }
}

// ---------------------------------------------------------------------------

extern "C" void kernel_launch(void* const* d_in, const int* in_sizes, int n_in,
                              void* d_out, int out_size, void* d_ws, size_t ws_size,
                              hipStream_t stream)
{
    const float* alpha     = (const float*)d_in[0];
    const float* beta_raw  = (const float*)d_in[1];
    const float* gamma_raw = (const float*)d_in[2];
    const float* T_mask    = (const float*)d_in[3];
    const float* VE_mask   = (const float*)d_in[4];
    const float* EV_mask   = (const float*)d_in[5];
    const float* accept    = (const float*)d_in[6];
    const float* range_m   = (const float*)d_in[7];
    const int*   prefixes  = (const int*)d_in[8];
    const int*   suffix    = (const int*)d_in[9];
    const int*   states    = (const int*)d_in[10];
    float* out = (float*)d_out;

    // ---- carve workspace (~490 MB) ----
    char* w = (char*)d_ws;
    auto alloc = [&](size_t bytes) -> void* {
        void* p = (void*)w;
        w += (bytes + 255) & ~(size_t)255;
        return p;
    };
    float* rowmax   = (float*)alloc((size_t)Hd * 4);
    float* rowStat  = (float*)alloc((size_t)Hd * 4);
    float* gammaW   = (float*)alloc((size_t)Hd * 4);
    float* colmaxB  = (float*)alloc((size_t)Vd * 4);
    float* BmT      = (float*)alloc((size_t)Hd * 4);
    float* alphaExp = (float*)alloc((size_t)Hd * Hd * 4);
    h16*   aTpk     = (h16*)  alloc((size_t)Hd * Hd * 2);   // B-op [N=H][K=H]
    h16*   Ptwpk    = (h16*)  alloc((size_t)Hd * Vd * 2);   // B-op [N=H][K=V]
    h16*   expBpk   = (h16*)  alloc((size_t)Vd * Hd * 2);   // B-op [N=V][K=H]
    h16*   TmaskA   = (h16*)  alloc((size_t)Ed * Vd * 2);   // A-op [E][V]
    h16*   TmaskB   = (h16*)  alloc((size_t)Vd * Ed * 2);   // B-op [N=V][K=E]
    h16*   EVpk     = (h16*)  alloc((size_t)Ed * Sd * 2);   // A-op [E][S]
    h16*   VEpk     = (h16*)  alloc((size_t)Sd * Ed * 2);   // A-op [S][E]
    float* Tw       = (float*)alloc((size_t)Ed * Hd * 4);
    float* Acache   = (float*)alloc((size_t)Bd * Hd * 4);
    float* ysuf     = (float*)alloc((size_t)Hd * 4);
    float* ysufmax  = (float*)alloc(256);
    float* Bm1      = (float*)alloc((size_t)Hd * 4);
    float* Bm2      = (float*)alloc((size_t)Hd * 4);
    float* Am3      = (float*)alloc((size_t)Sd * 4);
    h16*   E1       = (h16*)  alloc((size_t)Hd * Sd * 2);   // B-op [N=H][K=S]
    h16*   E2       = (h16*)  alloc((size_t)Hd * Ed * 2);   // B-op [N=H][K=E]
    h16*   E3       = (h16*)  alloc((size_t)Sd * Hd * 2);   // A-op [S][H]
    float* Z1       = (float*)alloc((size_t)Ed * Hd * 4);
    float* Z2       = (float*)alloc((size_t)Sd * Hd * 4);
    float* Cbuf     = (float*)alloc((size_t)Tp1d * Sd * Hd * 4);
    float* Cr       = (float*)alloc((size_t)Bd * Sd * Hd * 4);
    float* AmC      = (float*)alloc((size_t)BSd * 4);
    h16*   expCpk   = (h16*)  alloc((size_t)BSd * Hd * 2);  // A-op [BS][H]
    float* lp       = (float*)alloc((size_t)BSd * Vd * 4);
    h16*   M3pk     = (h16*)  alloc((size_t)BSd * Ed * 2);  // A-op [BS][E]
    float* mW       = (float*)alloc((size_t)BSd * Vd * 4);
    float* expA     = (float*)alloc((size_t)Bd * Hd * 4);
    float* rmaxA    = (float*)alloc((size_t)Bd * 4);

    // ---- stage 1: stats + packed operand conversion ----
    k_row_stats<<<Hd, 256, 0, stream>>>(beta_raw, rowmax, rowStat);
    k_gamma<<<1, 256, 0, stream>>>(gamma_raw, gammaW);
    k_alpha<<<Hd, 256, 0, stream>>>(alpha, alphaExp, aTpk);
    k_colmaxB<<<(Vd + 255) / 256, 256, 0, stream>>>(beta_raw, rowStat, colmaxB);
    k_bmT<<<(Hd + 255) / 256, 256, 0, stream>>>(rowmax, rowStat, BmT);
    {
        const size_t n = (size_t)Hd * Vd;
        k_packPtw<<<(unsigned)((n + 255) / 256), 256, 0, stream>>>(beta_raw, rowmax, Ptwpk);
    }
    k_packExpB<<<dim3(Hd / 32, Vd / 32), dim3(32, 8), 0, stream>>>(beta_raw, rowStat, colmaxB, expBpk);
    k_packMaskT<<<dim3(Ed / 32, Vd / 32), dim3(32, 8), 0, stream>>>(T_mask, TmaskB);
    k_packA<<<(Ed * Vd + 255) / 256, 256, 0, stream>>>(T_mask, TmaskA, Ed, Vd);
    k_packA<<<(Ed * Sd + 255) / 256, 256, 0, stream>>>(EV_mask, EVpk, Ed, Sd);
    k_packA<<<(Sd * Ed + 255) / 256, 256, 0, stream>>>(VE_mask, VEpk, Sd, Ed);

    // ---- stage 2: T_weights GEMM (E x V) @ (V x H), K = 32000 ----
    k_gemm_wmma<<<dim3(Hd / 64, Ed / 64), 128, 0, stream>>>(
        TmaskA, Ptwpk, Tw, Ed, Hd, Vd, nullptr, BmT, EPI_LOG_N2N);

    // ---- stage 3: sequential HMM scans ----
    k_prefix<<<Bd, 1024, 0, stream>>>(alphaExp, beta_raw, rowStat, gammaW, prefixes, Acache);
    k_suffix<<<1, 1024, 0, stream>>>(aTpk, beta_raw, rowStat, suffix, ysuf, ysufmax);

    // ---- stage 4: y0 + DFA scan (T+1 slabs of [S,H]) ----
    k_y0prep<<<Sd, 256, 0, stream>>>(ysuf, ysufmax, accept, Am3, E3);
    k_gemm_wmma<<<dim3(Hd / 64, Sd / 64), 128, 0, stream>>>(
        E3, aTpk, Cbuf, Sd, Hd, Hd, Am3, nullptr, EPI_LOG);
    for (int t = 1; t < Tp1d; ++t) {
        const float* Cin = Cbuf + (size_t)(t - 1) * Sd * Hd;
        float* Cout      = Cbuf + (size_t)t * Sd * Hd;
        k_colmax_exp<<<Hd / 256, 256, 0, stream>>>(Cin, nullptr, Bm1, E1, Sd);
        k_gemm_wmma<<<dim3(Hd / 64, Ed / 64), 128, 0, stream>>>(
            EVpk, E1, Z1, Ed, Hd, Sd, nullptr, Bm1, EPI_LOG_N2N);
        k_colmax_exp<<<Hd / 256, 256, 0, stream>>>(Z1, Tw, Bm2, E2, Ed);
        k_gemm_wmma<<<dim3(Hd / 64, Sd / 64), 128, 0, stream>>>(
            VEpk, E2, Z2, Sd, Hd, Ed, nullptr, Bm2, EPI_LOG_N2N);
        k_rowmax_exp<<<Sd, 256, 0, stream>>>(Z2, Am3, E3);
        k_gemm_wmma<<<dim3(Hd / 64, Sd / 64), 128, 0, stream>>>(
            E3, aTpk, Cout, Sd, Hd, Hd, Am3, nullptr, EPI_LOG);
    }

    // ---- stage 5: range aggregation + main logits ----
    k_range<<<(Sd * Hd) / 256, 256, 0, stream>>>(Cbuf, range_m, Cr);
    k_cfull_prep<<<BSd, 256, 0, stream>>>(Acache, Cr, AmC, expCpk);
    k_gemm_wmma<<<dim3(Vd / 64, BSd / 64), 128, 0, stream>>>(
        expCpk, expBpk, lp, BSd, Vd, Hd, AmC, colmaxB, EPI_LOG);
    k_m3<<<(BSd * Ed + 255) / 256, 256, 0, stream>>>(VE_mask, EV_mask, states, M3pk);
    k_gemm_wmma<<<dim3(Vd / 64, BSd / 64), 128, 0, stream>>>(
        M3pk, TmaskB, mW, BSd, Vd, Ed, nullptr, nullptr, EPI_PLAIN);
    k_final<<<(Bd * Vd + 255) / 256, 256, 0, stream>>>(lp, mW, out);

    // ---- stage 6: logits_ = matmul_log(A, beta) ----
    k_prepA<<<Bd, 256, 0, stream>>>(Acache, rmaxA, expA);
    k_logits2<<<(Vd + 255) / 256, 256, 0, stream>>>(expA, rmaxA, beta_raw, rowStat,
                                                    colmaxB, out + (size_t)Bd * Vd);
    (void)in_sizes; (void)n_in; (void)out_size; (void)ws_size;
}
